// LSTM_6786048328562
// MI455X (gfx1250) — matrix operand order repaired and checked by reading, not compiled, not run on hardware
//
#include <hip/hip_runtime.h>
#include <cstdint>

typedef __attribute__((ext_vector_type(16))) __bf16        v16bf;
typedef __attribute__((ext_vector_type(8)))  float         v8f;
typedef __attribute__((ext_vector_type(8)))  unsigned int  v8u;
typedef __attribute__((ext_vector_type(4)))  unsigned int  u32x4;
typedef __attribute__((ext_vector_type(8)))  int           i32x8;
typedef __attribute__((ext_vector_type(4)))  int           i32x4;

#define WMMA_BF16(A, B, C) \
  __builtin_amdgcn_wmma_f32_16x16x32_bf16(false, (A), false, (B), (short)0, (C), false, false)

#if defined(__has_builtin)
#if __has_builtin(__builtin_amdgcn_tensor_load_to_lds)
#define HAVE_TDM 1
#else
#define HAVE_TDM 0
#endif
#else
#define HAVE_TDM 0
#endif

// ---------- bf16 helpers (round-to-nearest-even) ----------
__device__ __forceinline__ unsigned short f2bf(float f) {
  unsigned int u = __float_as_uint(f);
  return (unsigned short)((u + 0x7FFFu + ((u >> 16) & 1u)) >> 16);
}
__device__ __forceinline__ float bf2f(unsigned short h) {
  return __uint_as_float(((unsigned int)h) << 16);
}
__device__ __forceinline__ float sigm(float x) { return 1.0f / (1.0f + __expf(-x)); }

// ---------- WMMA fragment loaders ----------
// A fragment: 16x32 bf16 (MxK). lanes 0-15 -> M=lane, K pairs {0..3,8..11};
// lanes 16-31 -> M=lane-16, K pairs {4..7,12..15}. (pair == one dword)
__device__ __forceinline__ v16bf load_a_frag(const unsigned short* __restrict__ base,
                                             size_t row, int lda, int k0, int lane) {
  const unsigned int* p = (const unsigned int*)(base + row * (size_t)lda + k0);
  int kbp = (lane & 16) ? 4 : 0;
  v8u r;
#pragma unroll
  for (int j = 0; j < 8; ++j) {
    int pj = kbp + j + ((j >= 4) ? 4 : 0);
    r[j] = p[pj];
  }
  return __builtin_bit_cast(v16bf, r);
}

// B fragment: 32x16 bf16 (KxN) for D = A*B where B^T is row-major [N][K].
// lanes 0-15 -> N=lane, K pairs {0..3,8..11}; lanes 16-31 -> N=lane-16, K pairs {4..7,12..15}.
__device__ __forceinline__ v16bf load_b_frag(const unsigned short* __restrict__ base,
                                             size_t col, int ldb, int k0, int lane) {
  const unsigned int* p = (const unsigned int*)(base + col * (size_t)ldb + k0);
  int kbp = (lane & 16) ? 4 : 0;
  v8u r;
#pragma unroll
  for (int j = 0; j < 8; ++j) r[j] = p[kbp + j + ((j >= 4) ? 4 : 0)];
  return __builtin_bit_cast(v16bf, r);
}

// ---------- TDM: async 2D-tile DMA global -> LDS (Tensor Data Mover) ----------
// Descriptor packing per CDNA5 ISA 8.3/8.4 (D# groups). 2D tensor: groups 2/3 zero.
__device__ __forceinline__ void tdm_load_2d_f32(unsigned lds_addr, const void* gaddr,
                                                unsigned tensor_w, unsigned tensor_h,
                                                unsigned long long row_stride,
                                                unsigned tile_w, unsigned tile_h) {
#if HAVE_TDM
  unsigned long long ga = (unsigned long long)(size_t)gaddr;
  u32x4 g0;
  g0[0] = 1u;                                                 // count=1, user mode
  g0[1] = lds_addr;                                           // lds_addr[31:0]
  g0[2] = (unsigned)(ga & 0xFFFFFFFFu);                       // global_addr[31:0]
  g0[3] = (unsigned)((ga >> 32) & 0x01FFFFFFu) | (2u << 30);  // global_addr[56:32], type=2
  i32x8 g1;
  g1[0] = (int)(2u << 16);                                    // data_size=2 (4 bytes)
  g1[1] = (int)((tensor_w & 0xFFFFu) << 16);                  // tensor_dim0[15:0]
  g1[2] = (int)(((tensor_w >> 16) & 0xFFFFu) | ((tensor_h & 0xFFFFu) << 16));
  g1[3] = (int)(((tensor_h >> 16) & 0xFFFFu) | ((tile_w & 0xFFFFu) << 16));  // tile_dim0
  g1[4] = (int)(tile_h & 0xFFFFu);                            // tile_dim1 (tile_dim2=0)
  g1[5] = (int)(unsigned)(row_stride & 0xFFFFFFFFu);          // tensor_dim0_stride[31:0]
  g1[6] = (int)(unsigned)((row_stride >> 32) & 0xFFFFu);      // tensor_dim0_stride[47:32]
  g1[7] = 0;
  i32x4 z4 = {0, 0, 0, 0};
#if __clang_major__ >= 23
  i32x8 z8 = {0, 0, 0, 0, 0, 0, 0, 0};
  __builtin_amdgcn_tensor_load_to_lds(g0, g1, z4, z4, z8, 0);
#else
  __builtin_amdgcn_tensor_load_to_lds(g0, g1, z4, z4, 0);
#endif
#else
  (void)lds_addr; (void)gaddr; (void)tensor_w; (void)tensor_h;
  (void)row_stride; (void)tile_w; (void)tile_h;
#endif
}

// ---------- fp32 -> (bf16 hi, bf16 lo) split ----------
__global__ void split_bf16(const float* __restrict__ src, unsigned short* __restrict__ hi,
                           unsigned short* __restrict__ lo, int n) {
  int i = blockIdx.x * blockDim.x + threadIdx.x;
  if (i < n) {
    float f = src[i];
    unsigned short h = f2bf(f);
    hi[i] = h;
    lo[i] = f2bf(f - bf2f(h));
  }
}

__global__ void init_state(float* __restrict__ c, unsigned short* __restrict__ h_hi,
                           unsigned short* __restrict__ h_lo, int n) {
  int i = blockIdx.x * blockDim.x + threadIdx.x;
  if (i < n) { c[i] = 0.0f; h_hi[i] = 0; h_lo[i] = 0; }
}

// ---------- Phase 1: xg[M1][N1] = x[M1][K] @ W[N1][K]^T + (b_ih + b_hh) ----------
// One wave computes a 16x64 block (4 tiles): A(hi,lo) fragments amortized over 12 WMMAs/chunk.
__global__ void __launch_bounds__(128) xw_gemm_wmma(
    const unsigned short* __restrict__ a_hi, const unsigned short* __restrict__ a_lo,
    const unsigned short* __restrict__ b_hi, const unsigned short* __restrict__ b_lo,
    const float* __restrict__ bias_ih, const float* __restrict__ bias_hh,
    float* __restrict__ xg) {
  constexpr int K = 1024, N = 4096;
  int wave = blockIdx.x * 4 + (threadIdx.x >> 5);
  int lane = threadIdx.x & 31;
  int tn4 = wave & 63;   // N/64 = 64 super-tiles
  int tm  = wave >> 6;   // M/16 = 1024 tiles
  size_t arow  = (size_t)tm * 16 + (lane & 15);
  size_t bcol0 = (size_t)tn4 * 64 + (lane & 15);

  v8f acc[4] = {};
  for (int k0 = 0; k0 < K; k0 += 32) {
    v16bf ah = load_a_frag(a_hi, arow, K, k0, lane);
    v16bf al = load_a_frag(a_lo, arow, K, k0, lane);
#pragma unroll
    for (int s = 0; s < 4; ++s) {
      v16bf bh = load_b_frag(b_hi, bcol0 + s * 16, K, k0, lane);
      v16bf bl = load_b_frag(b_lo, bcol0 + s * 16, K, k0, lane);
      acc[s] = WMMA_BF16(al, bh, acc[s]);
      acc[s] = WMMA_BF16(ah, bl, acc[s]);
      acc[s] = WMMA_BF16(ah, bh, acc[s]);
    }
  }
  // C/D layout: lanes 0-15 -> N=lane, M=v; lanes 16-31 -> N=lane-16, M=8+v.
  int mbase = tm * 16 + ((lane & 16) ? 8 : 0);
#pragma unroll
  for (int s = 0; s < 4; ++s) {
    int n = tn4 * 64 + s * 16 + (lane & 15);
    float bias = bias_ih[n] + bias_hh[n];
#pragma unroll
    for (int v = 0; v < 8; ++v) xg[(size_t)(mbase + v) * N + n] = acc[s][v] + bias;
  }
}

// ---------- Phase 2: one timestep.  g = xg[t] + h_{t-1} @ U^T, then gate math ----------
// 4 waves split K; wave 0 additionally TDM-prefetches the 4 gate tiles of xg[t] into LDS,
// overlapped with the WMMA K-loop, fenced by s_wait_tensorcnt before the fused epilogue.
__global__ void __launch_bounds__(128) lstm_step_wmma(
    const unsigned short* __restrict__ u_hi, const unsigned short* __restrict__ u_lo,
    const unsigned short* __restrict__ hr_hi, const unsigned short* __restrict__ hr_lo,
    unsigned short* __restrict__ hw_hi, unsigned short* __restrict__ hw_lo,
    float* __restrict__ c_state, const float* __restrict__ xg_t,
    float* __restrict__ out_t, float* __restrict__ h_n, float* __restrict__ c_n,
    int is_last) {
  constexpr int H = 1024;
  __shared__ float red[4 * 4 * 8 * 32];   // [wave][gate][v][lane] = 16 KB
  __shared__ float lds_xg[4 * 16 * 16];   // [gate][row][col] tiles of xg[t] = 4 KB

  int wave = threadIdx.x >> 5;
  int lane = threadIdx.x & 31;
  int btile = blockIdx.x & 1;    // B=32 -> 2 row tiles
  int jtile = blockIdx.x >> 1;   // H/16 = 64 col tiles
  size_t arow = (size_t)btile * 16 + (lane & 15);
  int jcol = jtile * 16 + (lane & 15);

  if (wave == 0) {
#if HAVE_TDM
    unsigned lds_base = (unsigned)(size_t)(void*)&lds_xg[0];
#pragma unroll
    for (int g = 0; g < 4; ++g) {
      const float* tile = xg_t + (size_t)(btile * 16) * (4 * H) + g * H + jtile * 16;
      tdm_load_2d_f32(lds_base + g * 1024u, tile, 4 * H, 32, 4 * H, 16, 16);
    }
#else
    for (int i = lane; i < 4 * 16 * 16; i += 32) {
      int g = i >> 8, r = (i >> 4) & 15, cc = i & 15;
      lds_xg[i] = xg_t[(size_t)(btile * 16 + r) * (4 * H) + g * H + jtile * 16 + cc];
    }
#endif
  }

  v8f acc[4] = {};
  int kbeg = wave * 256, kend = kbeg + 256;
  for (int k0 = kbeg; k0 < kend; k0 += 32) {
    v16bf ah = load_a_frag(hr_hi, arow, H, k0, lane);
    v16bf al = load_a_frag(hr_lo, arow, H, k0, lane);
#pragma unroll
    for (int g = 0; g < 4; ++g) {
      v16bf bh = load_b_frag(u_hi, (size_t)g * H + jcol, H, k0, lane);
      v16bf bl = load_b_frag(u_lo, (size_t)g * H + jcol, H, k0, lane);
      acc[g] = WMMA_BF16(al, bh, acc[g]);
      acc[g] = WMMA_BF16(ah, bl, acc[g]);
      acc[g] = WMMA_BF16(ah, bh, acc[g]);
    }
  }
#pragma unroll
  for (int g = 0; g < 4; ++g)
#pragma unroll
    for (int v = 0; v < 8; ++v) red[((wave * 4 + g) * 8 + v) * 32 + lane] = acc[g][v];
  __syncthreads();

  if (threadIdx.x < 32) {
#if HAVE_TDM
    __builtin_amdgcn_s_wait_tensorcnt((short)0);  // xg tiles landed in LDS
#endif
    int mofs = (lane & 16) ? 8 : 0;
    int j = jtile * 16 + (lane & 15);
#pragma unroll
    for (int v = 0; v < 8; ++v) {
      int ml = mofs + v;                 // row within 16-row tile
      int b = btile * 16 + ml;
      int cc = lane & 15;
      float g0 = lds_xg[0 * 256 + ml * 16 + cc];
      float g1 = lds_xg[1 * 256 + ml * 16 + cc];
      float g2 = lds_xg[2 * 256 + ml * 16 + cc];
      float g3 = lds_xg[3 * 256 + ml * 16 + cc];
#pragma unroll
      for (int w = 0; w < 4; ++w) {
        g0 += red[((w * 4 + 0) * 8 + v) * 32 + lane];
        g1 += red[((w * 4 + 1) * 8 + v) * 32 + lane];
        g2 += red[((w * 4 + 2) * 8 + v) * 32 + lane];
        g3 += red[((w * 4 + 3) * 8 + v) * 32 + lane];
      }
      float it = sigm(g0), ft = sigm(g1), ch = tanhf(g2), ot = sigm(g3);
      size_t idx = (size_t)b * H + j;
      float cv = ft * c_state[idx] + it * ch;
      float hv = ot * tanhf(cv);
      c_state[idx] = cv;
      out_t[idx] = hv;
      unsigned short hh = f2bf(hv);
      hw_hi[idx] = hh;
      hw_lo[idx] = f2bf(hv - bf2f(hh));
      if (is_last) { h_n[idx] = hv; c_n[idx] = cv; }
    }
  }
}

extern "C" void kernel_launch(void* const* d_in, const int* in_sizes, int n_in,
                              void* d_out, int out_size, void* d_ws, size_t ws_size,
                              hipStream_t stream) {
  (void)in_sizes; (void)n_in; (void)out_size; (void)ws_size;
  constexpr int T = 512, B = 32, I = 1024, H = 1024;
  constexpr size_t M1 = (size_t)T * B;   // 16384
  constexpr size_t N1 = 4 * (size_t)H;   // 4096

  const float* x    = (const float*)d_in[0];
  const float* W    = (const float*)d_in[1];
  const float* U    = (const float*)d_in[2];
  const float* b_ih = (const float*)d_in[3];
  const float* b_hh = (const float*)d_in[4];
  float* out = (float*)d_out;

  char* ws = (char*)d_ws;
  size_t off = 0;
  auto carve = [&](size_t bytes) -> char* {
    char* p = ws + off;
    off += (bytes + 255) & ~(size_t)255;
    return p;
  };
  unsigned short* x_hi = (unsigned short*)carve(M1 * I * 2);
  unsigned short* x_lo = (unsigned short*)carve(M1 * I * 2);
  unsigned short* w_hi = (unsigned short*)carve(N1 * I * 2);
  unsigned short* w_lo = (unsigned short*)carve(N1 * I * 2);
  unsigned short* u_hi = (unsigned short*)carve(N1 * H * 2);
  unsigned short* u_lo = (unsigned short*)carve(N1 * H * 2);
  float* xg      = (float*)carve(M1 * N1 * 4);
  float* c_state = (float*)carve((size_t)B * H * 4);
  unsigned short* h_hi[2] = {(unsigned short*)carve((size_t)B * H * 2),
                             (unsigned short*)carve((size_t)B * H * 2)};
  unsigned short* h_lo[2] = {(unsigned short*)carve((size_t)B * H * 2),
                             (unsigned short*)carve((size_t)B * H * 2)};

  // fp32 -> bf16 hi/lo splits
  { int n = (int)(M1 * I); split_bf16<<<(n + 255) / 256, 256, 0, stream>>>(x, x_hi, x_lo, n); }
  { int n = (int)(N1 * I); split_bf16<<<(n + 255) / 256, 256, 0, stream>>>(W, w_hi, w_lo, n); }
  { int n = (int)(N1 * H); split_bf16<<<(n + 255) / 256, 256, 0, stream>>>(U, u_hi, u_lo, n); }
  { int n = B * H; init_state<<<(n + 255) / 256, 256, 0, stream>>>(c_state, h_hi[0], h_lo[0], n); }

  // Phase 1: xg = x @ W^T + (b_ih + b_hh); 16x64 per wave
  {
    int waves = (int)((M1 / 16) * (N1 / 64));  // 65536
    xw_gemm_wmma<<<waves / 4, 128, 0, stream>>>(x_hi, x_lo, w_hi, w_lo, b_ih, b_hh, xg);
  }

  // Phase 2: 512 dependent steps; graph-captured launches give the global sync.
  float* h_n = out + (size_t)T * B * H;
  float* c_n = h_n + (size_t)B * H;
  for (int t = 0; t < T; ++t) {
    int r = t & 1, w = r ^ 1;
    lstm_step_wmma<<<128, 128, 0, stream>>>(u_hi, u_lo, h_hi[r], h_lo[r], h_hi[w], h_lo[w],
                                            c_state, xg + (size_t)t * B * N1,
                                            out + (size_t)t * B * H, h_n, c_n,
                                            (t == T - 1) ? 1 : 0);
  }
}
